// EncoderRNN_10874857194114
// MI455X (gfx1250) — matrix operand order repaired and checked by reading, not compiled, not run on hardware
//
#include <hip/hip_runtime.h>
#include <hip/hip_bf16.h>
#include <math.h>

#define B_   256
#define T_   128
#define E_   512
#define H_   512
#define G4H  2048  // 4*H

typedef __bf16 bf16_t;
typedef __attribute__((ext_vector_type(8)))  __bf16 bf16x8;
typedef __attribute__((ext_vector_type(16))) __bf16 bf16x16;
typedef __attribute__((ext_vector_type(8)))  float  f32x8;

__device__ __forceinline__ float sigmoidf_(float x) {
    return 1.0f / (1.0f + expf(-x));
}

// ---------------------------------------------------------------------------
// Embedding gather (f32 -> bf16), row layout X[t*B + b][E]; also emits mask.
// ---------------------------------------------------------------------------
__global__ void embed_mask_kernel(const int* __restrict__ inputs,
                                  const int* __restrict__ lengths,
                                  const float* __restrict__ emb,
                                  bf16_t* __restrict__ X,
                                  float* __restrict__ mask_out) {
    const int row = blockIdx.x;        // row = t*B + b
    const int t = row / B_;
    const int b = row % B_;
    const int v = inputs[b * T_ + t];
    const float* src = emb + (size_t)v * E_;
    bf16_t* dst = X + (size_t)row * E_;
    for (int i = threadIdx.x; i < E_; i += blockDim.x)
        dst[i] = (bf16_t)src[i];
    if (threadIdx.x == 0)
        mask_out[b * T_ + t] = (t < lengths[b]) ? 1.0f : 0.0f;
}

// ---------------------------------------------------------------------------
// Weight conversion f32 -> bf16 for W_ih and W_hh (each 2048x512).
// ---------------------------------------------------------------------------
__global__ void convert_w_kernel(const float* __restrict__ Wih,
                                 const float* __restrict__ Whh,
                                 bf16_t* __restrict__ Wih_b,
                                 bf16_t* __restrict__ Whh_b) {
    const int n = G4H * E_;
    int idx = blockIdx.x * blockDim.x + threadIdx.x;
    if (idx < n) {
        Wih_b[idx] = (bf16_t)Wih[idx];
    } else {
        int j = idx - n;
        if (j < n) Whh_b[j] = (bf16_t)Whh[j];
    }
}

// ---------------------------------------------------------------------------
// Input projection: gates_x[row][n] = X[row,:] . W_ih[n,:] + b_ih[n] + b_hh[n]
// rows = T*B = 32768, n = 0..2047.  Wave tile: 16 rows x 64 cols (4 WMMA accs).
// Mapping: the 8 waves of a workgroup share the SAME 64-col weight strip
// (identical B-tile addresses -> WGP$ hits) and cover 8 different M tiles.
// ---------------------------------------------------------------------------
__global__ __launch_bounds__(256)
void input_proj_kernel(const bf16_t* __restrict__ X,
                       const bf16_t* __restrict__ W,
                       const float* __restrict__ b_ih,
                       const float* __restrict__ b_hh,
                       float* __restrict__ gx) {
    const int lane   = threadIdx.x & 31;
    const int wv     = threadIdx.x >> 5;
    const int id     = blockIdx.x * 8 + wv;   // 0..65535
    const int mtile  = id & 2047;             // varies across waves in a block
    const int nstrip = id >> 11;              // 0..31, shared within a block
    const int l15    = lane & 15;
    const int khalf  = lane >> 4;
    const int mrow   = mtile * 16 + l15;      // A: lanes 0-15 M=lane, 16-31 M=lane-16
    const int ncol   = nstrip * 64 + l15;     // B: lane%16 = N

    f32x8 acc0 = {}, acc1 = {}, acc2 = {}, acc3 = {};
    const bf16_t* arow = X + (size_t)mrow * E_;

#pragma unroll
    for (int k0 = 0; k0 < E_; k0 += 32) {
        // A 16x32 bf16: lane loads 16 contiguous K at khalf*16
        bf16x16 a = *(const bf16x16*)(arow + k0 + khalf * 16);
        // B 32x16 bf16: lane (N=l15) loads 16 contiguous K at khalf*16
        const int kb = k0 + khalf * 16;
        bf16x16 b0 = *(const bf16x16*)(W + (size_t)(ncol +  0) * E_ + kb);
        bf16x16 b1 = *(const bf16x16*)(W + (size_t)(ncol + 16) * E_ + kb);
        bf16x16 b2 = *(const bf16x16*)(W + (size_t)(ncol + 32) * E_ + kb);
        bf16x16 b3 = *(const bf16x16*)(W + (size_t)(ncol + 48) * E_ + kb);
        acc0 = __builtin_amdgcn_wmma_f32_16x16x32_bf16(false, a, false, b0, (short)0, acc0, false, false);
        acc1 = __builtin_amdgcn_wmma_f32_16x16x32_bf16(false, a, false, b1, (short)0, acc1, false, false);
        acc2 = __builtin_amdgcn_wmma_f32_16x16x32_bf16(false, a, false, b2, (short)0, acc2, false, false);
        acc3 = __builtin_amdgcn_wmma_f32_16x16x32_bf16(false, a, false, b3, (short)0, acc3, false, false);
    }

    // C/D layout: VGPR r -> M = (lane>>4)*8 + r ; N = lane&15
    const int rbase = mtile * 16 + khalf * 8;
    const int n0 = nstrip * 64 + l15;
#pragma unroll
    for (int r = 0; r < 8; ++r) {
        const int row = rbase + r;
        float* dst = gx + (size_t)row * G4H + n0;
        dst[ 0] = acc0[r] + b_ih[n0 +  0] + b_hh[n0 +  0];
        dst[16] = acc1[r] + b_ih[n0 + 16] + b_hh[n0 + 16];
        dst[32] = acc2[r] + b_ih[n0 + 32] + b_hh[n0 + 32];
        dst[48] = acc3[r] + b_ih[n0 + 48] + b_hh[n0 + 48];
    }
}

// ---------------------------------------------------------------------------
// One LSTM step (both directions via blockIdx.y). Wave tile: 16 rows (batch)
// x 16 hidden cols, with all 4 gate tiles (n = g*512 + h) accumulated by the
// same wave -> fully fused cell update.
// Mapping: the 8 waves of a workgroup share the SAME hidden-column tile
// (identical W_hh B-tile addresses -> WGP$ hits) and cover 8 batch tiles.
// ---------------------------------------------------------------------------
__global__ __launch_bounds__(256)
void lstm_step_kernel(const bf16_t* __restrict__ Whh,
                      const float* __restrict__ gx,
                      const int* __restrict__ lengths,
                      const bf16_t* __restrict__ h_in_bf,   // [2][B][H] (read)
                      bf16_t* __restrict__ h_out_bf,        // [2][B][H] (write)
                      float* __restrict__ h32,              // [2][B][H]
                      float* __restrict__ c32,              // [2][B][H]
                      float* __restrict__ out,              // [B][T][2H]
                      int step) {
    const int dir   = blockIdx.y;
    const int t_eff = dir ? (T_ - 1 - step) : step;
    const int lane  = threadIdx.x & 31;
    const int wv    = threadIdx.x >> 5;
    const int id    = blockIdx.x * 8 + wv;  // 0..511
    const int mtile = id & 15;              // varies across waves in a block
    const int htile = id >> 4;              // 0..31, shared within a block
    const int l15   = lane & 15;
    const int khalf = lane >> 4;
    const int mrow  = mtile * 16 + l15;
    const int hc    = htile * 16 + l15;

    const bf16_t* hbase = h_in_bf + (size_t)dir * B_ * H_;
    const bf16_t* arow  = hbase + (size_t)mrow * H_;

    f32x8 ai = {}, af = {}, ag = {}, ao = {};
#pragma unroll
    for (int k0 = 0; k0 < H_; k0 += 32) {
        bf16x16 a = *(const bf16x16*)(arow + k0 + khalf * 16);
        const int kb = k0 + khalf * 16;
        bf16x16 bi = *(const bf16x16*)(Whh + (size_t)(0 * H_ + hc) * H_ + kb);
        bf16x16 bf = *(const bf16x16*)(Whh + (size_t)(1 * H_ + hc) * H_ + kb);
        bf16x16 bg = *(const bf16x16*)(Whh + (size_t)(2 * H_ + hc) * H_ + kb);
        bf16x16 bo = *(const bf16x16*)(Whh + (size_t)(3 * H_ + hc) * H_ + kb);
        ai = __builtin_amdgcn_wmma_f32_16x16x32_bf16(false, a, false, bi, (short)0, ai, false, false);
        af = __builtin_amdgcn_wmma_f32_16x16x32_bf16(false, a, false, bf, (short)0, af, false, false);
        ag = __builtin_amdgcn_wmma_f32_16x16x32_bf16(false, a, false, bg, (short)0, ag, false, false);
        ao = __builtin_amdgcn_wmma_f32_16x16x32_bf16(false, a, false, bo, (short)0, ao, false, false);
    }

    // Fused cell update.  C/D layout: VGPR r -> batch row (lane>>4)*8 + r.
    const int rm = khalf * 8;
#pragma unroll
    for (int r = 0; r < 8; ++r) {
        const int brow = mtile * 16 + rm + r;
        const float* gxr = gx + ((size_t)t_eff * B_ + brow) * G4H;
        const float iv = sigmoidf_(ai[r] + gxr[0 * H_ + hc]);
        const float fv = sigmoidf_(af[r] + gxr[1 * H_ + hc]);
        const float gv = tanhf    (ag[r] + gxr[2 * H_ + hc]);
        const float ov = sigmoidf_(ao[r] + gxr[3 * H_ + hc]);

        const size_t sidx = (size_t)dir * B_ * H_ + (size_t)brow * H_ + hc;
        const float c_old = c32[sidx];
        const float h_old = h32[sidx];
        const float c_new = fv * c_old + iv * gv;
        const float h_new = ov * tanhf(c_new);

        const bool m = t_eff < lengths[brow];
        const float hb = m ? h_new : h_old;
        const float cb = m ? c_new : c_old;

        c32[sidx] = cb;
        h32[sidx] = hb;
        h_out_bf[sidx] = (bf16_t)hb;
        out[(size_t)brow * (T_ * 2 * H_) + (size_t)t_eff * (2 * H_) + dir * H_ + hc] = hb;
    }
}

// ---------------------------------------------------------------------------
// Emit final ht/ct.
// ---------------------------------------------------------------------------
__global__ void final_state_kernel(const float* __restrict__ h32,
                                   const float* __restrict__ c32,
                                   float* __restrict__ out_ht,
                                   float* __restrict__ out_ct) {
    int idx = blockIdx.x * blockDim.x + threadIdx.x;   // over 2*B*H
    if (idx >= 2 * B_ * H_) return;
    const int dir = idx / (B_ * H_);
    const int rem = idx % (B_ * H_);
    const int b = rem / H_;
    const int h = rem % H_;
    out_ht[b * (2 * H_) + dir * H_ + h] = h32[idx];
    out_ct[b * (2 * H_) + dir * H_ + h] = c32[idx];
}

// ---------------------------------------------------------------------------
extern "C" void kernel_launch(void* const* d_in, const int* in_sizes, int n_in,
                              void* d_out, int out_size, void* d_ws, size_t ws_size,
                              hipStream_t stream) {
    const int*   inputs  = (const int*)  d_in[0];
    const int*   lengths = (const int*)  d_in[1];
    const float* emb     = (const float*)d_in[2];
    const float* W_ih    = (const float*)d_in[3];
    const float* W_hh    = (const float*)d_in[4];
    const float* b_ih    = (const float*)d_in[5];
    const float* b_hh    = (const float*)d_in[6];

    float* out = (float*)d_out;
    float* out_main = out;                                   // [B][T][2H]
    float* out_ht   = out + (size_t)B_ * T_ * 2 * H_;        // [B][2H]
    float* out_ct   = out_ht + (size_t)B_ * 2 * H_;          // [B][2H]
    float* out_mask = out_ct + (size_t)B_ * 2 * H_;          // [B][T]

    // Workspace layout (all offsets 1KB-aligned).
    char* ws = (char*)d_ws;
    const size_t off_X   = 0;                                  // 33,554,432 B
    const size_t off_Wih = off_X   + (size_t)T_ * B_ * E_ * 2; //  2,097,152 B
    const size_t off_Whh = off_Wih + (size_t)G4H * E_ * 2;     //  2,097,152 B
    const size_t off_gx  = off_Whh + (size_t)G4H * H_ * 2;     // 268,435,456 B
    const size_t off_h32 = off_gx  + (size_t)T_ * B_ * G4H * 4;
    const size_t off_c32 = off_h32 + (size_t)2 * B_ * H_ * 4;
    const size_t off_hbf = off_c32 + (size_t)2 * B_ * H_ * 4;  // 2 buffers bf16

    bf16_t* X     = (bf16_t*)(ws + off_X);
    bf16_t* Wih_b = (bf16_t*)(ws + off_Wih);
    bf16_t* Whh_b = (bf16_t*)(ws + off_Whh);
    float*  gx    = (float*) (ws + off_gx);
    float*  h32   = (float*) (ws + off_h32);
    float*  c32   = (float*) (ws + off_c32);
    bf16_t* hbf   = (bf16_t*)(ws + off_hbf);   // [2][2*B*H]

    // Zero h32, c32 and both bf16 h buffers (contiguous region).
    hipMemsetAsync(ws + off_h32,
                   0,
                   (size_t)2 * B_ * H_ * 4 * 2 + (size_t)2 * (2 * B_ * H_) * 2,
                   stream);

    embed_mask_kernel<<<T_ * B_, 256, 0, stream>>>(inputs, lengths, emb, X, out_mask);
    convert_w_kernel<<<(2 * G4H * E_) / 256, 256, 0, stream>>>(W_ih, W_hh, Wih_b, Whh_b);

    // gates_x = X @ W_ih^T + b_ih + b_hh   (shared by both directions)
    input_proj_kernel<<<8192, 256, 0, stream>>>(X, Wih_b, b_ih, b_hh, gx);

    // Sequential recurrence: 128 fused GEMM+cell steps, both directions.
    const size_t hbuf_elems = (size_t)2 * B_ * H_;
    for (int s = 0; s < T_; ++s) {
        bf16_t* h_in  = hbf + (size_t)(s & 1) * hbuf_elems;
        bf16_t* h_out = hbf + (size_t)((s + 1) & 1) * hbuf_elems;
        lstm_step_kernel<<<dim3(64, 2), 256, 0, stream>>>(
            Whh_b, gx, lengths, h_in, h_out, h32, c32, out_main, s);
    }

    final_state_kernel<<<(2 * B_ * H_) / 256, 256, 0, stream>>>(h32, c32, out_ht, out_ct);
}
